// MoELayer_91164975824951
// MI455X (gfx1250) — matrix operand order repaired and checked by reading, not compiled, not run on hardware
//
#include <hip/hip_runtime.h>
#include <hip/hip_bf16.h>
#include <cstdint>

typedef __attribute__((ext_vector_type(16))) __bf16 v16bf;
typedef __attribute__((ext_vector_type(8)))  float  v8f;

#define N_TOK 8192
#define HDIM  1024
#define FDIM  4096
#define EDIM  6

// GEMM tiling: workgroup 128(M) x 256(N), K-stage 64, 8 waves as 2(M) x 4(N),
// each wave owns a 64x64 tile = 4x4 WMMA C tiles.
#define BM 128
#define BN 256
#define BK 64

static __device__ __forceinline__ uint16_t f32_to_bf16(float f) {
    uint32_t u = __float_as_uint(f);
    if ((u & 0x7F800000u) == 0x7F800000u) return (uint16_t)((u | 0x00400000u) >> 16);
    return (uint16_t)((u + 0x7FFFu + ((u >> 16) & 1u)) >> 16);
}

// ---------------- gating: one wave (32 lanes) per token -------------------
__global__ void gate_kernel(const float* __restrict__ x, const float* __restrict__ Wg,
                            const float* __restrict__ bg, uint16_t* __restrict__ xb,
                            float* __restrict__ wf) {
    const int wave = threadIdx.x >> 5;
    const int lane = threadIdx.x & 31;
    const int n = blockIdx.x * 8 + wave;
    const float* xr = x + (size_t)n * HDIM;
    uint16_t*    xo = xb + (size_t)n * HDIM;
    float a0=0,a1=0,a2=0,a3=0,a4=0,a5=0;
    #pragma unroll 4
    for (int i = 0; i < HDIM/32; ++i) {
        int h = lane + 32*i;
        float xv = xr[h];
        xo[h] = f32_to_bf16(xv);
        const float* wr = Wg + (size_t)h * EDIM;
        a0 += xv*wr[0]; a1 += xv*wr[1]; a2 += xv*wr[2];
        a3 += xv*wr[3]; a4 += xv*wr[4]; a5 += xv*wr[5];
    }
    #pragma unroll
    for (int off = 16; off >= 1; off >>= 1) {
        a0 += __shfl_xor(a0, off, 32); a1 += __shfl_xor(a1, off, 32);
        a2 += __shfl_xor(a2, off, 32); a3 += __shfl_xor(a3, off, 32);
        a4 += __shfl_xor(a4, off, 32); a5 += __shfl_xor(a5, off, 32);
    }
    if (lane == 0) {
        float g[6] = {a0+bg[0], a1+bg[1], a2+bg[2], a3+bg[3], a4+bg[4], a5+bg[5]};
        int i0 = 0;
        #pragma unroll
        for (int e = 1; e < 6; ++e) if (g[e] > g[i0]) i0 = e;   // lowest-index tie-break
        int i1 = -1;
        #pragma unroll
        for (int e = 0; e < 6; ++e) if (e != i0 && (i1 < 0 || g[e] > g[i1])) i1 = e;
        float e1 = __expf(g[i1] - g[i0]);
        float inv = 1.0f / (1.0f + e1);
        float w0 = inv, w1 = e1 * inv;
        #pragma unroll
        for (int e = 0; e < 6; ++e)
            wf[(size_t)n*EDIM + e] = (e == i0) ? w0 : (e == i1) ? w1 : 0.0f;
    }
}

// -------------- out = sum_e w[n,e] * b2[e,:]  (bias epilogue) -------------
__global__ void out_init(const float* __restrict__ wf, const float* __restrict__ b2,
                         float* __restrict__ out) {
    int i = blockIdx.x * blockDim.x + threadIdx.x;      // N*H exact
    int n = i >> 10, h = i & 1023;
    const float* w = wf + (size_t)n * EDIM;
    float acc = 0.0f;
    #pragma unroll
    for (int e = 0; e < 6; ++e) acc += w[e] * b2[(size_t)e*HDIM + h];
    out[i] = acc;
}

// -------- fp32 [R][C] -> bf16 transposed [C][R], batched over z -----------
__global__ void transpose_to_bf16(const float* __restrict__ src, uint16_t* __restrict__ dst,
                                  int R, int C) {
    __shared__ float tile[32][33];
    const float* s = src + (size_t)blockIdx.z * R * C;
    uint16_t*    d = dst + (size_t)blockIdx.z * R * C;
    int c0 = blockIdx.x * 32, r0 = blockIdx.y * 32;
    int tx = threadIdx.x, ty = threadIdx.y;             // (32,8)
    #pragma unroll
    for (int i = 0; i < 4; ++i)
        tile[ty + 8*i][tx] = s[(size_t)(r0 + ty + 8*i) * C + c0 + tx];
    __syncthreads();
    #pragma unroll
    for (int i = 0; i < 4; ++i)
        d[(size_t)(c0 + ty + 8*i) * R + r0 + tx] = f32_to_bf16(tile[tx][ty + 8*i]);
}

// -------------------- shared BMxBNxK bf16 WMMA core -----------------------
// A: [M][K] bf16 row-major.  Bt: [Nout][K] bf16 (B stored transposed).
// Holds the 4 B fragments across the mb loop and streams A fragments, so the
// peak register set is acc(128) + B(32) + A(8) + addressing, no spills.
__device__ __forceinline__ void gemm_core(const uint16_t* __restrict__ A,
                                          const uint16_t* __restrict__ Bt,
                                          int K, int m0, int n0,
                                          uint16_t* As, uint16_t* Bs, v8f acc[4][4]) {
    const int t    = threadIdx.x;
    const int lane = t & 31;
    const int wave = t >> 5;
    const int wm   = wave >> 2;                         // 0..1 (M)
    const int wn   = wave & 3;                          // 0..3 (N)
    const int g    = lane >> 4;                         // K-half select
    const int ln   = lane & 15;

    for (int kk = 0; kk < K; kk += BK) {
        __syncthreads();
        // cooperative fill: A 128x64 (1024 chunks), B 256x64 (2048 chunks)
        #pragma unroll
        for (int i = 0; i < 4; ++i) {
            int ch = t + i*256, row = ch >> 3, c = ch & 7;
            *(uint4*)&As[row*BK + c*8] = *(const uint4*)&A[(size_t)(m0+row)*K + kk + c*8];
        }
        #pragma unroll
        for (int i = 0; i < 8; ++i) {
            int ch = t + i*256, row = ch >> 3, c = ch & 7;
            *(uint4*)&Bs[row*BK + c*8] = *(const uint4*)&Bt[(size_t)(n0+row)*K + kk + c*8];
        }
        if (kk + BK < K) {                              // prefetch next stage into caches
            __builtin_prefetch(&A [(size_t)(m0 + (t>>1))*K + kk + BK + (t&1)*32], 0, 1);
            __builtin_prefetch(&Bt[(size_t)(n0 + t)*K + kk + BK], 0, 1);
        }
        __syncthreads();

        #pragma unroll
        for (int kc = 0; kc < BK; kc += 32) {
            union { uint4 u[2]; v16bf v; } b[4];
            #pragma unroll
            for (int nb = 0; nb < 4; ++nb) {
                const uint4* rowB = (const uint4*)&Bs[(wn*64 + nb*16 + ln)*BK + kc];
                b[nb].u[0] = rowB[2*g];                 // K = 16g .. 16g+7
                b[nb].u[1] = rowB[2*g + 1];             // K = 16g+8 .. 16g+15
            }
            #pragma unroll
            for (int mb = 0; mb < 4; ++mb) {
                union { uint4 u[2]; v16bf v; } a;
                const uint4* rowA = (const uint4*)&As[(wm*64 + mb*16 + ln)*BK + kc];
                a.u[0] = rowA[2*g];                     // K = 16g .. 16g+7
                a.u[1] = rowA[2*g + 1];                 // K = 16g+8 .. 16g+15
                #pragma unroll
                for (int nb = 0; nb < 4; ++nb)
                    acc[mb][nb] = __builtin_amdgcn_wmma_f32_16x16x32_bf16(
                        false, a.v, false, b[nb].v, (short)0, acc[mb][nb], false, false);
            }
        }
    }
}

// --------- GEMM1: h = gelu(xb @ W1[e] + b1[e]) -> bf16 hbuf [N][F] --------
__global__ void __launch_bounds__(256, 1)
gemm1_gelu(const uint16_t* __restrict__ xb, const uint16_t* __restrict__ W1t,
           const float* __restrict__ b1, uint16_t* __restrict__ hbuf, int e) {
    __shared__ uint16_t As[BM*BK], Bs[BN*BK];
    const int n0 = blockIdx.x * BN;                     // over F
    const int m0 = blockIdx.y * BM;                     // over tokens
    v8f acc[4][4] = {};
    gemm_core(xb, W1t + (size_t)e*FDIM*HDIM, HDIM, m0, n0, As, Bs, acc);
    const int lane = threadIdx.x & 31, wave = threadIdx.x >> 5;
    const int wm = wave >> 2, wn = wave & 3, g = lane >> 4, ln = lane & 15;
    #pragma unroll
    for (int nb = 0; nb < 4; ++nb) {
        int f = n0 + wn*64 + nb*16 + ln;
        float bias = b1[(size_t)e*FDIM + f];
        #pragma unroll
        for (int mb = 0; mb < 4; ++mb) {
            #pragma unroll
            for (int v = 0; v < 8; ++v) {
                int m = m0 + wm*64 + mb*16 + v + 8*g;
                float val = acc[mb][nb][v] + bias;
                val = 0.5f * val * (1.0f + erff(val * 0.7071067811865476f)); // exact GELU
                hbuf[(size_t)m*FDIM + f] = f32_to_bf16(val);
            }
        }
    }
}

// ----- GEMM2: out[n,h] += w[n,e] * (hbuf @ W2[e])[n,h]  (b2 pre-added) ----
__global__ void __launch_bounds__(256, 1)
gemm2_acc(const uint16_t* __restrict__ hbuf, const uint16_t* __restrict__ W2t,
          const float* __restrict__ wf, float* __restrict__ out, int e) {
    __shared__ uint16_t As[BM*BK], Bs[BN*BK];
    const int n0 = blockIdx.x * BN;                     // over H
    const int m0 = blockIdx.y * BM;                     // over tokens
    v8f acc[4][4] = {};
    gemm_core(hbuf, W2t + (size_t)e*HDIM*FDIM, FDIM, m0, n0, As, Bs, acc);
    const int lane = threadIdx.x & 31, wave = threadIdx.x >> 5;
    const int wm = wave >> 2, wn = wave & 3, g = lane >> 4, ln = lane & 15;
    #pragma unroll
    for (int mb = 0; mb < 4; ++mb) {
        #pragma unroll
        for (int v = 0; v < 8; ++v) {
            int m = m0 + wm*64 + mb*16 + v + 8*g;
            float wt = wf[(size_t)m*EDIM + e];
            if (wt != 0.0f) {
                #pragma unroll
                for (int nb = 0; nb < 4; ++nb) {
                    int h = n0 + wn*64 + nb*16 + ln;
                    size_t o = (size_t)m*HDIM + h;
                    out[o] += wt * acc[mb][nb][v];
                }
            }
        }
    }
}

extern "C" void kernel_launch(void* const* d_in, const int* in_sizes, int n_in,
                              void* d_out, int out_size, void* d_ws, size_t ws_size,
                              hipStream_t stream) {
    const float* x  = (const float*)d_in[0];
    const float* Wg = (const float*)d_in[1];
    const float* bg = (const float*)d_in[2];
    const float* W1 = (const float*)d_in[3];
    const float* b1 = (const float*)d_in[4];
    const float* W2 = (const float*)d_in[5];
    const float* b2 = (const float*)d_in[6];
    float* out = (float*)d_out;

    char* ws = (char*)d_ws;
    uint16_t* xb   = (uint16_t*)ws;  ws += (size_t)N_TOK*HDIM*2;        // 16 MB
    float*    wf   = (float*)ws;     ws += (size_t)N_TOK*EDIM*4;        // 192 KB
    uint16_t* W1t  = (uint16_t*)ws;  ws += (size_t)EDIM*FDIM*HDIM*2;    // 48 MB
    uint16_t* W2t  = (uint16_t*)ws;  ws += (size_t)EDIM*HDIM*FDIM*2;    // 48 MB
    uint16_t* hbuf = (uint16_t*)ws;                                     // 64 MB

    gate_kernel<<<N_TOK/8, 256, 0, stream>>>(x, Wg, bg, xb, wf);
    out_init<<<(N_TOK*HDIM)/256, 256, 0, stream>>>(wf, b2, out);
    // W1[e]: [H][F] -> W1t[e]: [F][H]
    transpose_to_bf16<<<dim3(FDIM/32, HDIM/32, EDIM), dim3(32,8), 0, stream>>>(W1, W1t, HDIM, FDIM);
    // W2[e]: [F][H] -> W2t[e]: [H][F]
    transpose_to_bf16<<<dim3(HDIM/32, FDIM/32, EDIM), dim3(32,8), 0, stream>>>(W2, W2t, FDIM, HDIM);
    for (int e = 0; e < EDIM; ++e) {
        gemm1_gelu<<<dim3(FDIM/BN, N_TOK/BM), 256, 0, stream>>>(xb, W1t, b1, hbuf, e);
        gemm2_acc <<<dim3(HDIM/BN, N_TOK/BM), 256, 0, stream>>>(hbuf, W2t, wf, out, e);
    }
}
